// discrete_net_model_171798692344
// MI455X (gfx1250) — matrix operand order repaired and checked by reading, not compiled, not run on hardware
//
#include <hip/hip_runtime.h>
#include <hip/hip_bf16.h>
#include <math.h>

typedef __attribute__((ext_vector_type(16))) _Float16 v16h;
typedef __attribute__((ext_vector_type(8)))  _Float16 v8h;
typedef __attribute__((ext_vector_type(8)))  float    v8f;

#define Bn   16
#define Tn   64
#define Sn   256
#define INn  7
#define Hn   64
#define NOn  16
#define TPn  4
#define Kpad 224          // 32 (x padded) + 64 (h) + 64 (h_after) + 64 (h_before)
#define Ngate 256         // 4*H
#define NT_STEPS 63       // T-1
#define TOUT 59           // T-1-TP
#define OUTC 17           // NO + SO
#define WPACK_ELEMS (7*16*32*16)   // ktiles * ntiles * lanes * 16 halves = 57344

__device__ __forceinline__ float sigmoidf_(float x) { return 1.0f / (1.0f + __expf(-x)); }

// ---------------------------------------------------------------- zero init
__global__ void zero_f32_kernel(float* __restrict__ p, int n) {
    int i = blockIdx.x * 256 + threadIdx.x;
    if (i < n) p[i] = 0.0f;
}

// ------------------------------------------------- pack Wg (f32) -> B-frag f16
// Wpack flat index = ((kt*16 + ntile)*32 + lane)*16 + j
// lane<16 : K = kt*32 + j,      N = ntile*16 + lane
// lane>=16: K = kt*32 + 16 + j, N = ntile*16 + lane-16
__global__ void pack_wg_kernel(const float* __restrict__ Wg, _Float16* __restrict__ Wpack) {
    int idx = blockIdx.x * 256 + threadIdx.x;
    if (idx >= WPACK_ELEMS) return;
    int j    = idx & 15;
    int lane = (idx >> 4) & 31;
    int nt   = (idx >> 9) & 15;
    int kt   = idx >> 13;
    int hi   = lane >> 4;
    int K = kt * 32 + hi * 16 + j;
    int n = nt * 16 + (lane & 15);
    int r = -1;                       // padded-K -> original Wg row
    if (K < 32)        { if (K < INn) r = K; }
    else if (K < 96)   r = 7   + (K - 32);    // h
    else if (K < 160)  r = 71  + (K - 96);    // h_after
    else               r = 135 + (K - 160);   // h_before
    float v = (r >= 0) ? Wg[r * Ngate + n] : 0.0f;
    Wpack[idx] = (_Float16)v;
}

// ---------------------------------------------------------------- one timestep
// grid: 256 blocks (one 16-row M tile each), 256 threads (8 waves)
__global__ __launch_bounds__(256) void step_kernel(
    const float* __restrict__ x,          // [B,T,S,IN]
    const _Float16* __restrict__ Wpack,
    const float* __restrict__ bg,         // [256]
    const float* __restrict__ Wn,         // [64,16]
    const float* __restrict__ bn,         // [16]
    const float* __restrict__ Ws,         // [64]
    const float* __restrict__ bs,         // [1]
    const float* __restrict__ h_in,       // [B,S,H]
    float* __restrict__ h_out,            // [B,S,H]
    float* __restrict__ c,                // [B,S,H]
    float* __restrict__ out,              // [B,59,S,17]
    int t)
{
    __shared__ __align__(16) _Float16 Alds[16 * Kpad];   // 7 KB
    __shared__ __align__(16) float    zlds[16 * Ngate];  // 16 KB
    __shared__ __align__(16) float    hlds[16 * Hn];     // 4 KB

    const int tid = threadIdx.x;
    const int blk = blockIdx.x;          // 0..255
    const int b   = blk >> 4;            // batch
    const int s0  = (blk & 15) * 16;     // first spatial row of this tile

    // ---- stage A (concat features, f16, K padded to 224) into LDS ----------
    for (int idx = tid; idx < 16 * Kpad; idx += 256) {
        int m = idx / Kpad;
        int k = idx - m * Kpad;
        int s = s0 + m;
        float v = 0.0f;
        if (k < INn) {
            v = x[((b * Tn + t) * Sn + s) * INn + k];
        } else if (k >= 32 && k < 96) {
            v = h_in[(b * Sn + s) * Hn + (k - 32)];
        } else if (k >= 96 && k < 160) {
            int sa = s + 1;
            v = (sa < Sn) ? h_in[(b * Sn + sa) * Hn + (k - 96)] : 0.0f;
        } else if (k >= 160) {
            int sb = s - 1;
            v = (sb >= 0) ? h_in[(b * Sn + sb) * Hn + (k - 160)] : 0.0f;
        }
        Alds[m * Kpad + k] = (_Float16)v;
    }
    __syncthreads();

    // ---- WMMA: 16 x 256 gate pre-activations, 7 K-tiles ---------------------
    const int wave = tid >> 5;
    const int lane = tid & 31;
    const int m16  = lane & 15;
    const int hi   = lane >> 4;

    for (int q = 0; q < 2; ++q) {
        const int nt = wave * 2 + q;                 // N tile 0..15
        float bias = bg[nt * 16 + m16];
        v8f acc;
        #pragma unroll
        for (int v = 0; v < 8; ++v) acc[v] = bias;

        #pragma unroll
        for (int kt = 0; kt < 7; ++kt) {
            // A fragment: lanes 0-15 K = kt*32 + {0..15}; lanes 16-31 +16
            const _Float16* ap = &Alds[m16 * Kpad + kt * 32 + hi * 16];
            v8h a0 = *(const v8h*)(ap);
            v8h a1 = *(const v8h*)(ap + 8);
            v16h a = __builtin_shufflevector(a0, a1,
                        0,1,2,3,4,5,6,7, 8,9,10,11,12,13,14,15);
            // B fragment: contiguous 16 halves per lane from packed Wg
            v16h bf = *(const v16h*)&Wpack[(((kt * 16 + nt) * 32) + lane) * 16];
            acc = __builtin_amdgcn_wmma_f32_16x16x32_f16(
                      false, a, false, bf, (short)0, acc, false, false);
        }
        #pragma unroll
        for (int v = 0; v < 8; ++v)
            zlds[(v + 8 * hi) * Ngate + nt * 16 + m16] = acc[v];
    }
    __syncthreads();

    // ---- LSTM elementwise update -------------------------------------------
    for (int idx = tid; idx < 16 * Hn; idx += 256) {
        int m = idx >> 6;
        int u = idx & 63;
        int s = s0 + m;
        float zi = zlds[m * Ngate + u];
        float zf = zlds[m * Ngate + 64 + u];
        float zg = zlds[m * Ngate + 128 + u];
        float zo = zlds[m * Ngate + 192 + u];
        int ci = (b * Sn + s) * Hn + u;
        float cn = sigmoidf_(zf) * c[ci] + sigmoidf_(zi) * tanhf(zg);
        float hn = sigmoidf_(zo) * tanhf(cn);
        c[ci] = cn;
        h_out[ci] = hn;
        hlds[m * Hn + u] = hn;
    }
    __syncthreads();

    // ---- output projection (only kept timesteps) ---------------------------
    if (t >= TPn) {
        for (int idx = tid; idx < 16 * OUTC; idx += 256) {
            int m = idx / OUTC;
            int j = idx - m * OUTC;
            int s = s0 + m;
            float acc;
            if (j < NOn) {
                acc = bn[j];
                #pragma unroll
                for (int u = 0; u < Hn; ++u) acc += hlds[m * Hn + u] * Wn[u * NOn + j];
            } else {
                acc = bs[0];
                #pragma unroll
                for (int u = 0; u < Hn; ++u) acc += hlds[m * Hn + u] * Ws[u];
            }
            out[(((b * TOUT) + (t - TPn)) * Sn + s) * OUTC + j] = acc;
        }
    }
}

// ---------------------------------------------------------------- delta_N
__global__ void delta_kernel(const float* __restrict__ x,
                             const float* __restrict__ out,
                             float* __restrict__ delta) {
    int idx = blockIdx.x * 256 + threadIdx.x;       // over B*TOUT*S
    if (idx >= Bn * TOUT * Sn) return;
    int s  = idx & (Sn - 1);
    int tt = (idx >> 8) % TOUT;
    int b  = idx / (TOUT * Sn);
    const float* opb = out + (size_t)((b * TOUT + tt) * Sn) * OUTC;

    float op[NOn], ip[NOn];
    #pragma unroll
    for (int o = 0; o < NOn; ++o) op[o] = opb[s * OUTC + o];
    if (s == 0) {
        int cls = (int)x[((b * Tn + (tt + TPn)) * Sn + 0) * INn + 0];
        #pragma unroll
        for (int o = 0; o < NOn; ++o) ip[o] = (o == cls) ? 1.0f : 0.0f;
    } else {
        #pragma unroll
        for (int o = 0; o < NOn; ++o) ip[o] = opb[(s - 1) * OUTC + o];
    }
    #pragma unroll
    for (int l = 0; l < 2 * NOn - 1; ++l) {
        float acc = 0.0f;
        #pragma unroll
        for (int o = 0; o < NOn; ++o) {
            int src = o + l - (NOn - 1);
            if (src >= 0 && src < NOn) acc += ip[o] * op[src];
        }
        delta[(size_t)idx * (2 * NOn - 1) + l] = acc;
    }
}

extern "C" void kernel_launch(void* const* d_in, const int* in_sizes, int n_in,
                              void* d_out, int out_size, void* d_ws, size_t ws_size,
                              hipStream_t stream) {
    const float* x  = (const float*)d_in[0];
    const float* Wg = (const float*)d_in[1];
    const float* bg = (const float*)d_in[2];
    const float* Wn = (const float*)d_in[3];
    const float* bn = (const float*)d_in[4];
    const float* Ws = (const float*)d_in[5];
    const float* bs = (const float*)d_in[6];
    float* out = (float*)d_out;

    char* ws = (char*)d_ws;
    const size_t HBYTES = (size_t)Bn * Sn * Hn * sizeof(float);   // 1 MB
    float* h0 = (float*)(ws);
    float* h1 = (float*)(ws + HBYTES);
    float* c  = (float*)(ws + 2 * HBYTES);
    _Float16* Wpack = (_Float16*)(ws + 3 * HBYTES);

    // zero h0,h1,c (contiguous)
    int nz = (int)(3 * HBYTES / sizeof(float));
    zero_f32_kernel<<<(nz + 255) / 256, 256, 0, stream>>>((float*)ws, nz);
    // pack weights into WMMA B-fragment layout (f16, K padded to 224)
    pack_wg_kernel<<<(WPACK_ELEMS + 255) / 256, 256, 0, stream>>>(Wg, Wpack);

    for (int t = 0; t < NT_STEPS; ++t) {
        const float* hin = (t & 1) ? h1 : h0;
        float* hout      = (t & 1) ? h0 : h1;
        step_kernel<<<256, 256, 0, stream>>>(x, Wpack, bg, Wn, bn, Ws, bs,
                                             hin, hout, c, out, t);
    }

    float* delta = out + (size_t)Bn * TOUT * Sn * OUTC;
    int nd = Bn * TOUT * Sn;
    delta_kernel<<<(nd + 255) / 256, 256, 0, stream>>>(x, out, delta);
}
